// VanillaStructureTokenEncoder_10290741641486
// MI455X (gfx1250) — compile-verified, matrix operand order checked by reading, not run on hardware
//
#include <hip/hip_runtime.h>
#include <math.h>

// ---------------- problem constants ----------------
#define D_MODEL   1024
#define NHEADS    128
#define KNN       16
#define LSEQ      512
#define NBATCH    2
#define NRES      (NBATCH*LSEQ)   // 1024 residues
#define NK        (NRES*KNN)      // 16384 token rows
#define DPROJ     1920            // 4*H*3 + H*3
#define FFN       4096
#define INV_SQRT3 0.57735026918962576f

// ---------------- WMMA types ----------------
typedef __attribute__((ext_vector_type(16))) __bf16       v16bf;
typedef __attribute__((ext_vector_type(8)))  float        v8f;
typedef __attribute__((ext_vector_type(4)))  unsigned int u32x4;
typedef __attribute__((ext_vector_type(4)))  int          v4i;

// pointer-to-int4 in global (AS1) / LDS (AS3) address spaces, per the
// builtin's prototype: (v4i AS1* gaddr, v4i AS3* lds, int offset, int cpol)
typedef v4i __attribute__((address_space(1)))* gv4ip;
typedef v4i __attribute__((address_space(3)))* lv4ip;

union FragU { v16bf v; u32x4 u[2]; };

// ---------------- GEMM tiling ----------------
#define BM   128
#define BN   128
#define BK   32
#define LDST 40           // LDS row stride (elements): 32 + 8 pad, keeps 16B align

// Stage one 16-byte chunk global -> LDS (async if the toolchain has the
// gfx1250 builtin, else a plain b128 copy behind the barrier).
__device__ __forceinline__ void cp_b128(const __bf16* gp, __bf16* sp) {
#if __has_builtin(__builtin_amdgcn_global_load_async_to_lds_b128)
  __builtin_amdgcn_global_load_async_to_lds_b128((gv4ip)gp, (lv4ip)sp, 0, 0);
#else
  *reinterpret_cast<u32x4*>(sp) = *reinterpret_cast<const u32x4*>(gp);
#endif
}

__device__ __forceinline__ void wait_async_lds() {
#if __has_builtin(__builtin_amdgcn_s_wait_asynccnt)
  __builtin_amdgcn_s_wait_asynccnt(0);
#else
  asm volatile("s_wait_asynccnt 0x0" ::: "memory");
#endif
}

// Fragment from an LDS tile (row-major rows x BK, stride LDST).
// lane 0-15 -> row, K = 0..7 & 16..23 ; lane 16-31 -> same rows, K = 8..15 & 24..31.
__device__ __forceinline__ v16bf lds_frag(const __bf16* s, int row0, int lane) {
  int row = row0 + (lane & 15);
  int kb  = (lane >> 4) << 3;
  const __bf16* p = s + row * LDST + kb;
  FragU f;
  f.u[0] = *reinterpret_cast<const u32x4*>(p);
  f.u[1] = *reinterpret_cast<const u32x4*>(p + 16);
  return f.v;
}

__device__ __forceinline__ void store_tile(const v8f& acc, int mfrag, int nfrag,
                                           const float* __restrict__ bias,
                                           const float* __restrict__ residual,
                                           float* Cf, __bf16* Cb,
                                           int Nn, int lane, int out_bf16) {
  int cm  = (lane >> 4) << 3;      // C layout: lanes 16-31 hold rows M=8..15
  int col = nfrag + (lane & 15);
  float bv = bias ? bias[col] : 0.0f;
#pragma unroll
  for (int i = 0; i < 8; ++i) {
    size_t idx = (size_t)(mfrag + cm + i) * Nn + col;
    float v = acc[i] + bv;
    if (residual) v += residual[idx];
    if (out_bf16) Cb[idx] = (__bf16)v;
    else          Cf[idx] = v;
  }
}

// C[M,Nn] = A[M,Kd] * W[Nn,Kd]^T (+bias) (+residual).
// Block 128x128, 8 waves as 4(M) x 2(N), wave tile 32x64 (8 WMMA / K-step).
// A/B K-tiles double-buffered in LDS, filled with async-to-LDS copies.
__global__ __launch_bounds__(256)
void gemm_bf16_kernel(const __bf16* __restrict__ A, const __bf16* __restrict__ W,
                      const float* __restrict__ bias, const float* __restrict__ residual,
                      float* Cf, __bf16* Cb, int M, int Nn, int Kd, int out_bf16) {
  __shared__ alignas(16) __bf16 sA[2][BM * LDST];
  __shared__ alignas(16) __bf16 sB[2][BN * LDST];
  int tid  = threadIdx.x;
  int lane = tid & 31;
  int wave = tid >> 5;                    // 0..7
  int mloc = (wave & 3) * 32;             // 4 waves in M
  int nloc = (wave >> 2) * 64;            // 2 waves in N
  long mbase = (long)blockIdx.x * BM;
  long nbase = (long)blockIdx.y * BN;

  v8f acc[2][4];
#pragma unroll
  for (int i = 0; i < 2; ++i)
#pragma unroll
    for (int j = 0; j < 4; ++j) acc[i][j] = {};

  // stage tile for K-step s into buffer buf: 512 chunks A + 512 chunks B,
  // 256 threads x (2 A + 2 B) 16-byte chunks.
  auto stage = [&](int buf, int k0) {
#pragma unroll
    for (int r = 0; r < 2; ++r) {
      int c   = tid + r * 256;
      int row = c >> 2;
      int kc  = (c & 3) << 3;
      cp_b128(A + (mbase + row) * Kd + k0 + kc, &sA[buf][row * LDST + kc]);
      cp_b128(W + (nbase + row) * Kd + k0 + kc, &sB[buf][row * LDST + kc]);
    }
  };

  stage(0, 0);
  wait_async_lds();
  __syncthreads();

  int nstep = Kd / BK;
  for (int s = 0; s < nstep; ++s) {
    int cur = s & 1;
    if (s + 1 < nstep) stage(cur ^ 1, (s + 1) * BK);

    v16bf af[2], bfr[4];
    af[0] = lds_frag(sA[cur], mloc,      lane);
    af[1] = lds_frag(sA[cur], mloc + 16, lane);
#pragma unroll
    for (int j = 0; j < 4; ++j) bfr[j] = lds_frag(sB[cur], nloc + j * 16, lane);
#pragma unroll
    for (int i = 0; i < 2; ++i)
#pragma unroll
      for (int j = 0; j < 4; ++j)
        acc[i][j] = __builtin_amdgcn_wmma_f32_16x16x32_bf16(
            false, af[i], false, bfr[j], (short)0, acc[i][j], false, false);

    wait_async_lds();
    __syncthreads();
  }

#pragma unroll
  for (int i = 0; i < 2; ++i)
#pragma unroll
    for (int j = 0; j < 4; ++j)
      store_tile(acc[i][j], (int)mbase + mloc + i * 16, (int)nbase + nloc + j * 16,
                 bias, residual, Cf, Cb, Nn, lane, out_bf16);
}

// ---------------- stage 0: frames ----------------
__device__ __forceinline__ void norm3(float* v) {
  float s = v[0]*v[0] + v[1]*v[1] + v[2]*v[2] + 1e-8f;
  float r = rsqrtf(s);
  v[0] *= r; v[1] *= r; v[2] *= r;
}

__global__ void frames_kernel(const float* __restrict__ coords,
                              float* __restrict__ rot, float* __restrict__ trans,
                              float* __restrict__ cam, int* __restrict__ fmask) {
  int idx = blockIdx.x * blockDim.x + threadIdx.x;
  if (idx >= NRES) return;
  const float* cr = coords + (size_t)idx * 9;
  float nn[3] = {cr[0], cr[1], cr[2]};
  float ca[3] = {cr[3], cr[4], cr[5]};
  float cc[3] = {cr[6], cr[7], cr[8]};
  int m = 1;
#pragma unroll
  for (int i = 0; i < 9; ++i) if (!isfinite(cr[i])) m = 0;
  float e0[3] = {cc[0]-ca[0], cc[1]-ca[1], cc[2]-ca[2]};
  norm3(e0);
  float v[3] = {nn[0]-ca[0], nn[1]-ca[1], nn[2]-ca[2]};
  float d = e0[0]*v[0] + e0[1]*v[1] + e0[2]*v[2];
  float e1[3] = {v[0]-e0[0]*d, v[1]-e0[1]*d, v[2]-e0[2]*d};
  norm3(e1);
  float e2[3] = {e0[1]*e1[2]-e0[2]*e1[1], e0[2]*e1[0]-e0[0]*e1[2], e0[0]*e1[1]-e0[1]*e1[0]};
  float* R = rot + (size_t)idx * 9;       // R[a][b] = component a of e_b
#pragma unroll
  for (int a = 0; a < 3; ++a) { R[a*3+0] = e0[a]; R[a*3+1] = e1[a]; R[a*3+2] = e2[a]; }
#pragma unroll
  for (int a = 0; a < 3; ++a) { trans[idx*3+a] = ca[a]; cam[idx*3+a] = m ? ca[a] : 0.0f; }
  fmask[idx] = m;
}

// ---------------- stage 1: per-query top-16 nearest ----------------
__global__ void topk_kernel(const float* __restrict__ cam, const int* __restrict__ fmask,
                            const int* __restrict__ seqid, const int* __restrict__ amask,
                            int* __restrict__ edges) {
  int q = blockIdx.x * blockDim.x + threadIdx.x;
  if (q >= NRES) return;
  int b = q / LSEQ;
  float qx = cam[q*3], qy = cam[q*3+1], qz = cam[q*3+2];
  int   mq = fmask[q], sq = seqid[q];
  float bd[KNN]; int bi[KNN];
#pragma unroll
  for (int k = 0; k < KNN; ++k) { bd[k] = 3.0e38f; bi[k] = 0; }
  for (int j = 0; j < LSEQ; ++j) {
    int jj = b * LSEQ + j;
    float dx = qx - cam[jj*3], dy = qy - cam[jj*3+1], dz = qz - cam[jj*3+2];
    float dist = sqrtf(dx*dx + dy*dy + dz*dz);
    int valid = mq && fmask[jj] && (sq == seqid[jj]) && (amask[jj] != 0);
    float dv = valid ? dist : 1.0e9f;
    if (dv < bd[KNN-1]) {
      int pos = KNN - 1;
      while (pos > 0 && dv < bd[pos-1]) { bd[pos] = bd[pos-1]; bi[pos] = bi[pos-1]; --pos; }
      bd[pos] = dv; bi[pos] = j;          // strict '<' keeps stable tie order
    }
  }
#pragma unroll
  for (int k = 0; k < KNN; ++k) edges[q*KNN + k] = bi[k];
}

// ---------------- stage 2: gathers ----------------
__global__ void gather_kernel(const int* __restrict__ edges,
                              const float* __restrict__ rot, const float* __restrict__ trans,
                              const int* __restrict__ seqid, const int* __restrict__ resid,
                              float* __restrict__ grot, float* __restrict__ gtrans,
                              int* __restrict__ gseq, int* __restrict__ gres) {
  int e = blockIdx.x * blockDim.x + threadIdx.x;
  if (e >= NK) return;
  int n = e / KNN, b = n / LSEQ;
  int src = b * LSEQ + edges[e];
#pragma unroll
  for (int t = 0; t < 9; ++t) grot[(size_t)e*9 + t] = rot[(size_t)src*9 + t];
#pragma unroll
  for (int t = 0; t < 3; ++t) gtrans[e*3 + t] = trans[src*3 + t];
  gseq[e] = seqid[src];
  gres[e] = resid[src];
}

__global__ __launch_bounds__(256)
void build_x_kernel(const int* __restrict__ gres, const float* __restrict__ rel_emb,
                    float* __restrict__ x) {
  int e = blockIdx.x;                       // token row (n*K + k)
  int r0 = gres[(e / KNN) * KNN];
  int diff = gres[e] - r0;
  diff = diff < -32 ? -32 : (diff > 32 ? 32 : diff);
  int row = diff + 33;
  for (int c = threadIdx.x; c < D_MODEL; c += 256)
    x[(size_t)e * D_MODEL + c] = rel_emb[(size_t)row * D_MODEL + c];
}

__global__ void build_bias_kernel(const int* __restrict__ gseq, float* __restrict__ biasb) {
  int n = blockIdx.x, t = threadIdx.x;      // 256 threads = 16x16
  int i = t >> 4, j = t & 15;
  biasb[(size_t)n*256 + t] = (gseq[n*KNN+i] == gseq[n*KNN+j]) ? 0.0f : -1.0e9f;
}

// ---------------- LayerNorm -> bf16 ----------------
__global__ __launch_bounds__(256)
void ln_kernel(const float* __restrict__ x, const float* __restrict__ w,
               const float* __restrict__ b, __bf16* __restrict__ out) {
  int row = blockIdx.x, tid = threadIdx.x;
  const float* xr = x + (size_t)row * D_MODEL;
  float v[4], s = 0.0f, sq = 0.0f;
#pragma unroll
  for (int i = 0; i < 4; ++i) { v[i] = xr[tid + i*256]; s += v[i]; sq += v[i]*v[i]; }
#pragma unroll
  for (int o = 16; o > 0; o >>= 1) { s += __shfl_down(s, o, 32); sq += __shfl_down(sq, o, 32); }
  __shared__ float ss[8], ssq[8];
  __shared__ float mu_s, inv_s;
  if ((tid & 31) == 0) { ss[tid>>5] = s; ssq[tid>>5] = sq; }
  __syncthreads();
  if (tid == 0) {
    float t = 0.0f, t2 = 0.0f;
#pragma unroll
    for (int i = 0; i < 8; ++i) { t += ss[i]; t2 += ssq[i]; }
    float mu = t / (float)D_MODEL;
    float var = t2 / (float)D_MODEL - mu*mu;
    mu_s = mu; inv_s = rsqrtf(var + 1e-5f);
  }
  __syncthreads();
#pragma unroll
  for (int i = 0; i < 4; ++i) {
    int c = tid + i*256;
    out[(size_t)row * D_MODEL + c] = (__bf16)((v[i] - mu_s) * inv_s * w[c] + b[c]);
  }
}

// ---------------- elementwise ----------------
__global__ void f2bf_kernel(const float* __restrict__ in, __bf16* __restrict__ out, long n) {
  long i = (long)blockIdx.x * blockDim.x + threadIdx.x;
  if (i < n) out[i] = (__bf16)in[i];
}

__global__ void gate_kernel(const __bf16* __restrict__ h, __bf16* __restrict__ act) {
  long i = (long)blockIdx.x * blockDim.x + threadIdx.x;
  if (i >= (long)NK * FFN) return;
  long m = i / FFN, c = i % FFN;
  float a = (float)h[m * (2*FFN) + c];
  float g = (float)h[m * (2*FFN) + FFN + c];
  float sil = a / (1.0f + expf(-a));
  act[i] = (__bf16)(sil * g);
}

__global__ __launch_bounds__(256)
void zmask_kernel(const float* __restrict__ x, const int* __restrict__ fmask,
                  __bf16* __restrict__ zbf) {
  int n = blockIdx.x;
  int m = fmask[n];
  const float* src = x + (size_t)(n * KNN) * D_MODEL;   // k = 0 row
  for (int c = threadIdx.x; c < D_MODEL; c += 256)
    zbf[(size_t)n * D_MODEL + c] = (__bf16)(m ? src[c] : 0.0f);
}

// ---------------- frame attention: 1 block / node, 1 thread / head ----------------
__device__ __forceinline__ void rot_apply(const float* R, const float* v, float* o) {
  o[0] = R[0]*v[0] + R[1]*v[1] + R[2]*v[2];
  o[1] = R[3]*v[0] + R[4]*v[1] + R[5]*v[2];
  o[2] = R[6]*v[0] + R[7]*v[1] + R[8]*v[2];
}

__global__ __launch_bounds__(128)
void attn_kernel(const float* __restrict__ p, const float* __restrict__ grot,
                 const float* __restrict__ gtrans, const float* __restrict__ biasb,
                 const float* __restrict__ rot_scale, const float* __restrict__ dist_scale,
                 __bf16* __restrict__ obf) {
  int n = blockIdx.x, h = threadIdx.x;
  __shared__ float sR[KNN*9];
  __shared__ float sT[KNN*3];
  __shared__ float sB[KNN*KNN];
  for (int t = h; t < KNN*9;   t += 128) sR[t] = grot[(size_t)n*KNN*9 + t];
  for (int t = h; t < KNN*3;   t += 128) sT[t] = gtrans[(size_t)n*KNN*3 + t];
  for (int t = h; t < KNN*KNN; t += 128) sB[t] = biasb[(size_t)n*KNN*KNN + t];
  __syncthreads();

  float kr[KNN][3], kd[KNN][3], vl[KNN][3], kn[KNN];
#pragma unroll
  for (int k = 0; k < KNN; ++k) {
    const float* pk = p + ((size_t)n*KNN + k) * DPROJ;
    float raw[3], tmp[3];
    raw[0]=pk[(NHEADS+h)*3+0]; raw[1]=pk[(NHEADS+h)*3+1]; raw[2]=pk[(NHEADS+h)*3+2];
    rot_apply(&sR[k*9], raw, kr[k]);                               // k_rot
    raw[0]=pk[(2*NHEADS+h)*3+0]; raw[1]=pk[(2*NHEADS+h)*3+1]; raw[2]=pk[(2*NHEADS+h)*3+2];
    rot_apply(&sR[k*9], raw, vl[k]);                               // val
    raw[0]=pk[9*NHEADS+(NHEADS+h)*3+0]; raw[1]=pk[9*NHEADS+(NHEADS+h)*3+1]; raw[2]=pk[9*NHEADS+(NHEADS+h)*3+2];
    rot_apply(&sR[k*9], raw, tmp);                                 // k_d
    kd[k][0]=tmp[0]+sT[k*3]; kd[k][1]=tmp[1]+sT[k*3+1]; kd[k][2]=tmp[2]+sT[k*3+2];
    kn[k] = kd[k][0]*kd[k][0] + kd[k][1]*kd[k][1] + kd[k][2]*kd[k][2];
  }
  float sp_rot  = log1pf(expf(rot_scale[h]));
  float sp_dist = log1pf(expf(dist_scale[h]));

  for (int i = 0; i < KNN; ++i) {
    const float* pi = p + ((size_t)n*KNN + i) * DPROJ;
    float raw[3], qr[3], qd[3], tmp[3];
    raw[0]=pi[h*3+0]; raw[1]=pi[h*3+1]; raw[2]=pi[h*3+2];
    rot_apply(&sR[i*9], raw, qr);                                  // q_rot
    raw[0]=pi[9*NHEADS+h*3+0]; raw[1]=pi[9*NHEADS+h*3+1]; raw[2]=pi[9*NHEADS+h*3+2];
    rot_apply(&sR[i*9], raw, tmp);                                 // q_d
    qd[0]=tmp[0]+sT[i*3]; qd[1]=tmp[1]+sT[i*3+1]; qd[2]=tmp[2]+sT[i*3+2];
    float qn = qd[0]*qd[0] + qd[1]*qd[1] + qd[2]*qd[2];

    float lg[KNN], mx = -3.0e38f;
#pragma unroll
    for (int j = 0; j < KNN; ++j) {
      float rt = (qr[0]*kr[j][0] + qr[1]*kr[j][1] + qr[2]*kr[j][2]) * INV_SQRT3;
      float qk =  qd[0]*kd[j][0] + qd[1]*kd[j][1] + qd[2]*kd[j][2];
      float dt = sqrtf(fmaxf(qn + kn[j] - 2.0f*qk, 1e-8f)) * INV_SQRT3;
      lg[j] = sp_rot*rt - sp_dist*dt + sB[i*KNN + j];
      mx = fmaxf(mx, lg[j]);
    }
    float s = 0.0f;
#pragma unroll
    for (int j = 0; j < KNN; ++j) { lg[j] = expf(lg[j] - mx); s += lg[j]; }
    float inv = 1.0f / s;
    float o0 = 0.0f, o1 = 0.0f, o2 = 0.0f;
#pragma unroll
    for (int j = 0; j < KNN; ++j) {
      float wj = lg[j] * inv;
      o0 += wj * vl[j][0]; o1 += wj * vl[j][1]; o2 += wj * vl[j][2];
    }
    // back-rotate by R_i^T: out_a = sum_b R[b][a] * o_b
    const float* R = &sR[i*9];
    float r0 = R[0]*o0 + R[3]*o1 + R[6]*o2;
    float r1 = R[1]*o0 + R[4]*o1 + R[7]*o2;
    float r2 = R[2]*o0 + R[5]*o1 + R[8]*o2;
    __bf16* dst = obf + ((size_t)n*KNN + i) * (3*NHEADS) + h*3;
    dst[0] = (__bf16)r0; dst[1] = (__bf16)r1; dst[2] = (__bf16)r2;
  }
}

// ---------------- host side ----------------
extern "C" void kernel_launch(void* const* d_in, const int* in_sizes, int n_in,
                              void* d_out, int out_size, void* d_ws, size_t ws_size,
                              hipStream_t stream) {
  const float* coords    = (const float*)d_in[0];
  const int*   amask     = (const int*)  d_in[1];
  const int*   seqid     = (const int*)  d_in[2];
  const int*   resid     = (const int*)  d_in[3];
  const float* rel_emb   = (const float*)d_in[4];
  const float* ln1_w     = (const float*)d_in[5];
  const float* ln1_b     = (const float*)d_in[6];
  const float* proj_w    = (const float*)d_in[7];
  const float* proj_b    = (const float*)d_in[8];
  const float* out_w     = (const float*)d_in[9];
  const float* out_b     = (const float*)d_in[10];
  const float* dist_sc   = (const float*)d_in[11];
  const float* rot_sc    = (const float*)d_in[12];
  const float* ln2_w     = (const float*)d_in[13];
  const float* ln2_b     = (const float*)d_in[14];
  const float* w1        = (const float*)d_in[15];
  const float* b1        = (const float*)d_in[16];
  const float* w2        = (const float*)d_in[17];
  const float* b2        = (const float*)d_in[18];
  const float* vq_w      = (const float*)d_in[19];
  const float* vq_b      = (const float*)d_in[20];

  char* ws = (char*)d_ws;
  size_t off = 0;
  auto alloc = [&](size_t bytes) -> void* {
    void* p = ws + off; off += (bytes + 255) & ~(size_t)255; return p;
  };
  float*  rotb   = (float*) alloc((size_t)NRES*9*4);
  float*  transb = (float*) alloc((size_t)NRES*3*4);
  float*  camb   = (float*) alloc((size_t)NRES*3*4);
  int*    fmaskb = (int*)   alloc((size_t)NRES*4);
  int*    edgesb = (int*)   alloc((size_t)NRES*KNN*4);
  float*  grotb  = (float*) alloc((size_t)NK*9*4);
  float*  gtrb   = (float*) alloc((size_t)NK*3*4);
  int*    gseqb  = (int*)   alloc((size_t)NK*4);
  int*    gresb  = (int*)   alloc((size_t)NK*4);
  float*  biasb  = (float*) alloc((size_t)NRES*KNN*KNN*4);
  float*  xb     = (float*) alloc((size_t)NK*D_MODEL*4);
  __bf16* nsb    = (__bf16*)alloc((size_t)NK*D_MODEL*2);
  float*  pb     = (float*) alloc((size_t)NK*DPROJ*4);
  __bf16* obfb   = (__bf16*)alloc((size_t)NK*3*NHEADS*2);
  __bf16* hb     = (__bf16*)alloc((size_t)NK*2*FFN*2);
  __bf16* actb   = (__bf16*)alloc((size_t)NK*FFN*2);
  __bf16* wbfb   = (__bf16*)alloc((size_t)(2*FFN)*D_MODEL*2);  // largest weight (w1)
  __bf16* zbfb   = (__bf16*)alloc((size_t)NRES*D_MODEL*2);

  // stage 0-2
  frames_kernel<<<(NRES+255)/256, 256, 0, stream>>>(coords, rotb, transb, camb, fmaskb);
  topk_kernel<<<(NRES+255)/256, 256, 0, stream>>>(camb, fmaskb, seqid, amask, edgesb);
  gather_kernel<<<(NK+255)/256, 256, 0, stream>>>(edgesb, rotb, transb, seqid, resid,
                                                  grotb, gtrb, gseqb, gresb);
  build_x_kernel<<<NK, 256, 0, stream>>>(gresb, rel_emb, xb);
  build_bias_kernel<<<NRES, 256, 0, stream>>>(gseqb, biasb);

  for (int l = 0; l < 2; ++l) {
    // ---- attention sub-block ----
    ln_kernel<<<NK, 256, 0, stream>>>(xb, ln1_w + l*D_MODEL, ln1_b + l*D_MODEL, nsb);
    long nw = (long)DPROJ * D_MODEL;
    f2bf_kernel<<<(nw+255)/256, 256, 0, stream>>>(proj_w + (size_t)l*nw, wbfb, nw);
    gemm_bf16_kernel<<<dim3(NK/BM, DPROJ/BN), 256, 0, stream>>>(
        nsb, wbfb, proj_b + (size_t)l*DPROJ, nullptr, pb, nullptr, NK, DPROJ, D_MODEL, 0);
    attn_kernel<<<NRES, 128, 0, stream>>>(pb, grotb, gtrb, biasb,
                                          rot_sc + l*NHEADS, dist_sc + l*NHEADS, obfb);
    nw = (long)D_MODEL * (3*NHEADS);
    f2bf_kernel<<<(nw+255)/256, 256, 0, stream>>>(out_w + (size_t)l*nw, wbfb, nw);
    gemm_bf16_kernel<<<dim3(NK/BM, D_MODEL/BN), 256, 0, stream>>>(
        obfb, wbfb, out_b + (size_t)l*D_MODEL, xb, xb, nullptr, NK, D_MODEL, 3*NHEADS, 0);

    // ---- FFN sub-block ----
    ln_kernel<<<NK, 256, 0, stream>>>(xb, ln2_w + l*D_MODEL, ln2_b + l*D_MODEL, nsb);
    nw = (long)(2*FFN) * D_MODEL;
    f2bf_kernel<<<(nw+255)/256, 256, 0, stream>>>(w1 + (size_t)l*nw, wbfb, nw);
    gemm_bf16_kernel<<<dim3(NK/BM, (2*FFN)/BN), 256, 0, stream>>>(
        nsb, wbfb, b1 + (size_t)l*2*FFN, nullptr, nullptr, hb, NK, 2*FFN, D_MODEL, 1);
    long ng = (long)NK * FFN;
    gate_kernel<<<(ng+255)/256, 256, 0, stream>>>(hb, actb);
    nw = (long)D_MODEL * FFN;
    f2bf_kernel<<<(nw+255)/256, 256, 0, stream>>>(w2 + (size_t)l*nw, wbfb, nw);
    gemm_bf16_kernel<<<dim3(NK/BM, D_MODEL/BN), 256, 0, stream>>>(
        actb, wbfb, b2 + (size_t)l*D_MODEL, xb, xb, nullptr, NK, D_MODEL, FFN, 0);
  }

  // ---- output head ----
  zmask_kernel<<<NRES, 256, 0, stream>>>(xb, fmaskb, zbfb);
  long nv = (long)128 * D_MODEL;
  f2bf_kernel<<<(nv+255)/256, 256, 0, stream>>>(vq_w, wbfb, nv);
  gemm_bf16_kernel<<<dim3(NRES/BM, 128/BN), 256, 0, stream>>>(
      zbfb, wbfb, vq_b, nullptr, (float*)d_out, nullptr, NRES, 128, D_MODEL, 0);
  (void)in_sizes; (void)n_in; (void)out_size; (void)ws_size;
}
